// Mixer2dTriU_61598420959800
// MI455X (gfx1250) — matrix operand order repaired and checked by reading, not compiled, not run on hardware
//
#include <hip/hip_runtime.h>
#include <math.h>

// B=64, T=512, C=512
#define TB 64
#define TT 512
#define TC 512
#define NTC (TT * TC)            // 262144 elems per batch
#define NTOT ((size_t)TB * NTC)  // 16777216 elems
#define LNEPS 1e-5f

typedef __attribute__((ext_vector_type(16))) __bf16 v16bf;
typedef __attribute__((ext_vector_type(8)))  __bf16 v8bf;
typedef __attribute__((ext_vector_type(8)))  float  v8f;
typedef __attribute__((ext_vector_type(4)))  unsigned int v4u;
typedef __attribute__((ext_vector_type(8)))  int v8i;
typedef __attribute__((ext_vector_type(4)))  int v4i;

// Triple-buffered tile stages: each stage = A(128x32 bf16, 8KB) + B(128x32, 8KB)
#define STAGE_BYTES 16384
#define STAGE_ELEMS 8192  // bf16 elems per stage
#define NSTAGE 3

// ---------------------------------------------------------------------------
// LayerNorm statistics over one batch's [T,C] slab (262144 floats).
// ---------------------------------------------------------------------------
__global__ __launch_bounds__(256) void ln_stats_kernel(const float* __restrict__ x,
                                                       float* __restrict__ mu,
                                                       float* __restrict__ rstd) {
  const int b = blockIdx.x;
  const float* p = x + (size_t)b * NTC;
  float s = 0.f, s2 = 0.f;
  for (int i = threadIdx.x; i < NTC; i += 256) {
    float v = p[i];
    s += v;
    s2 += v * v;
  }
  __shared__ float sh[256];
  __shared__ float sh2[256];
  sh[threadIdx.x] = s;
  sh2[threadIdx.x] = s2;
  __syncthreads();
  for (int st = 128; st > 0; st >>= 1) {
    if (threadIdx.x < st) {
      sh[threadIdx.x] += sh[threadIdx.x + st];
      sh2[threadIdx.x] += sh2[threadIdx.x + st];
    }
    __syncthreads();
  }
  if (threadIdx.x == 0) {
    const float inv = 1.0f / (float)NTC;
    float m = sh[0] * inv;
    float var = sh2[0] * inv - m * m;
    mu[b] = m;
    rstd[b] = rsqrtf(var + LNEPS);
  }
}

// ---------------------------------------------------------------------------
// tril(M) -> bf16   (512x512)
// ---------------------------------------------------------------------------
__global__ __launch_bounds__(256) void tril_to_bf16_kernel(const float* __restrict__ m,
                                                           __bf16* __restrict__ out) {
  int idx = blockIdx.x * 256 + threadIdx.x;
  if (idx < TT * TT) {
    int i = idx >> 9, j = idx & (TT - 1);
    out[idx] = (__bf16)((j <= i) ? m[idx] : 0.0f);
  }
}

__global__ __launch_bounds__(256) void to_bf16_kernel(const float* __restrict__ a,
                                                      __bf16* __restrict__ out, int n) {
  int idx = blockIdx.x * 256 + threadIdx.x;
  if (idx < n) out[idx] = (__bf16)a[idx];
}

// ---------------------------------------------------------------------------
// Apply LN1 and write TRANSPOSED bf16: x1t[b][c][t] = LN(x)[b][t][c]
// ---------------------------------------------------------------------------
__global__ __launch_bounds__(256) void ln1_apply_t_kernel(const float* __restrict__ x,
                                                          const float* __restrict__ w,
                                                          const float* __restrict__ bias,
                                                          const float* __restrict__ mu,
                                                          const float* __restrict__ rstd,
                                                          __bf16* __restrict__ xt) {
  __shared__ float tile[32][33];
  const int b = blockIdx.z;
  const int t0 = blockIdx.y << 5;
  const int c0 = blockIdx.x << 5;
  const float m = mu[b], rs = rstd[b];
  for (int r = threadIdx.y; r < 32; r += 8) {
    int t = t0 + r, c = c0 + threadIdx.x;
    size_t gi = ((size_t)b << 18) + ((size_t)t << 9) + c;
    int wi = (t << 9) + c;
    tile[r][threadIdx.x] = (x[gi] - m) * rs * w[wi] + bias[wi];
  }
  __syncthreads();
  for (int r = threadIdx.y; r < 32; r += 8) {
    int c = c0 + r, t = t0 + threadIdx.x;
    xt[((size_t)b << 18) + ((size_t)c << 9) + t] = (__bf16)tile[threadIdx.x][r];
  }
}

// ---------------------------------------------------------------------------
// Apply LN2 elementwise: x2f (f32, -> d_out) and x2b (bf16, GEMM operand)
// ---------------------------------------------------------------------------
__global__ __launch_bounds__(256) void ln2_apply_kernel(const float* __restrict__ u,
                                                        const float* __restrict__ w,
                                                        const float* __restrict__ bias,
                                                        const float* __restrict__ mu,
                                                        const float* __restrict__ rstd,
                                                        float* __restrict__ x2f,
                                                        __bf16* __restrict__ x2b) {
  size_t idx = (size_t)blockIdx.x * 256 + threadIdx.x;
  int b = (int)(idx >> 18);
  int tc = (int)(idx & (NTC - 1));
  float v = (u[idx] - mu[b]) * rstd[b] * w[tc] + bias[tc];
  x2f[idx] = v;
  x2b[idx] = (__bf16)v;
}

// ---------------------------------------------------------------------------
// Tensor Data Mover: DMA one 2-D tile (rows x cols bf16, row stride in elems)
// from global memory into LDS at byte offset lds_off. Descriptor packing per
// CDNA5 ISA 8.3/8.4: group0 {count=1, lds_addr, global_addr, type=2},
// group1 {data_size=2B, tensor dims, tile dims, dim0 stride}. 2-D only, so
// groups 2/3 are zero. Tiles are always in-bounds here.
// ---------------------------------------------------------------------------
__device__ __forceinline__ void tdm_load_tile_2d(unsigned lds_off,
                                                 const __bf16* gptr,
                                                 unsigned rows, unsigned cols,
                                                 unsigned row_stride) {
  unsigned long long ga = (unsigned long long)(uintptr_t)gptr;
  v4u g0;
  g0.x = 1u;                                   // count=1, user flags 0
  g0.y = lds_off;                              // LDS byte address
  g0.z = (unsigned)(ga & 0xFFFFFFFFu);         // global_addr[31:0]
  g0.w = (unsigned)((ga >> 32) & 0x01FFFFFFu) | (2u << 30);  // addr[56:32]|type=2
  const unsigned td0 = 512u, td1 = 512u;       // tensor dims (OOB guard only)
  v8i g1;
  g1[0] = (int)(1u << 16);                     // workgroup_mask=0, data_size=1 (2B)
  g1[1] = (int)((td0 & 0xFFFFu) << 16);        // tensor_dim0[15:0]
  g1[2] = (int)((td0 >> 16) | ((td1 & 0xFFFFu) << 16));   // td0 hi | td1 lo
  g1[3] = (int)((td1 >> 16) | (cols << 16));   // td1 hi | tile_dim0
  g1[4] = (int)rows;                           // tile_dim1 (tile_dim2=0)
  g1[5] = (int)row_stride;                     // tensor_dim0_stride[31:0]
  g1[6] = 0;                                   // stride hi | tensor_dim1_stride lo
  g1[7] = 0;
  v4i gz = {0, 0, 0, 0};
#if defined(__clang_major__) && (__clang_major__ >= 23)
  v8i gz8 = {0, 0, 0, 0, 0, 0, 0, 0};
  __builtin_amdgcn_tensor_load_to_lds(g0, g1, gz, gz, gz8, 0);
#else
  __builtin_amdgcn_tensor_load_to_lds(g0, g1, gz, gz, 0);
#endif
}

// ---------------------------------------------------------------------------
// Shared WMMA GEMM mainloop, TDM-fed, triple-buffered LDS, 1 barrier/K-step.
//   C[m0:m0+128, n0:n0+128] += A[m,k] * B^T[n,k],  K stepped 32 at a time.
//   Ag: row-major [M][512] bf16,  Bg: row-major [N][512] bf16 (i.e. B^T).
//   256 threads = 8 waves; wave (wrow,wcol) owns 32x64 = 2x4 16x16 tiles.
// Fragment layouts follow CDNA5 ISA 7.12.2 (wave32):
//   A 16x32: lane l16 -> row; K runs [hi*8, hi*8+8) and [16+hi*8, 16+hi*8+8)
//   B 32x16: lane l16 -> col; K runs [hi*8, hi*8+8) and [16+hi*8, 16+hi*8+8)
//   C/D    : vgpr r -> row r + 8*hi; lane l16 -> col
// ---------------------------------------------------------------------------
__device__ __forceinline__ void gemm_mainloop(const __bf16* __restrict__ Ag,
                                              const __bf16* __restrict__ Bg,
                                              int m0, int n0, int nk,
                                              __bf16* smem, v8f acc[2][4]) {
  const int tid = threadIdx.x;
  const int lane = tid & 31;
  const int wid = tid >> 5;
  const int wrow = wid & 3;
  const int wcol = wid >> 2;
  const int l16 = lane & 15;
  const int hi = lane >> 4;

  const __bf16* Abase = Ag + ((size_t)m0 << 9);
  const __bf16* Bbase = Bg + ((size_t)n0 << 9);
  // LDS byte offset of smem within the wave's LDS allocation (generic-pointer
  // low 32 bits are the LDS address).
  const unsigned lds0 = (unsigned)(uintptr_t)smem;

  if (wid == 0) {
    // Prologue: stages 0 and 1 in flight.
    tdm_load_tile_2d(lds0, Abase, 128, 32, 512);
    tdm_load_tile_2d(lds0 + 8192, Bbase, 128, 32, 512);
    if (nk > 1) {
      tdm_load_tile_2d(lds0 + STAGE_BYTES, Abase + 32, 128, 32, 512);
      tdm_load_tile_2d(lds0 + STAGE_BYTES + 8192, Bbase + 32, 128, 32, 512);
    }
  }

  for (int kb = 0; kb < nk; ++kb) {
    if (wid == 0) {
      // Stage kb complete when only stage kb+1's 2 TDM ops remain outstanding.
      if (kb + 1 < nk) __builtin_amdgcn_s_wait_tensorcnt(2);
      else             __builtin_amdgcn_s_wait_tensorcnt(0);
    }
    __syncthreads();  // stage kb visible; all waves done reading stage kb-1
    if (wid == 0 && kb + 2 < nk) {
      const unsigned q = (unsigned)((kb + 2) % NSTAGE) * STAGE_BYTES;
      const int koff = (kb + 2) << 5;
      tdm_load_tile_2d(lds0 + q, Abase + koff, 128, 32, 512);
      tdm_load_tile_2d(lds0 + q + 8192, Bbase + koff, 128, 32, 512);
    }
    const __bf16* As = smem + (size_t)(kb % NSTAGE) * STAGE_ELEMS;
    const __bf16* Bs = As + 4096;

    v16bf af[2], bfv[4];
#pragma unroll
    for (int fm = 0; fm < 2; ++fm) {
      const __bf16* ap = As + ((wrow << 5) + (fm << 4) + l16) * 32;
      v8bf r1 = *(const v8bf*)(ap + hi * 8);
      v8bf r2 = *(const v8bf*)(ap + 16 + hi * 8);
      af[fm] = __builtin_shufflevector(r1, r2, 0, 1, 2, 3, 4, 5, 6, 7, 8, 9, 10,
                                       11, 12, 13, 14, 15);
    }
#pragma unroll
    for (int fn = 0; fn < 4; ++fn) {
      const __bf16* bp = Bs + ((wcol << 6) + (fn << 4) + l16) * 32;
      v8bf q1 = *(const v8bf*)(bp + hi * 8);
      v8bf q2 = *(const v8bf*)(bp + 16 + hi * 8);
      bfv[fn] = __builtin_shufflevector(q1, q2, 0, 1, 2, 3, 4, 5, 6, 7, 8, 9, 10,
                                        11, 12, 13, 14, 15);
    }
#pragma unroll
    for (int fm = 0; fm < 2; ++fm)
#pragma unroll
      for (int fn = 0; fn < 4; ++fn)
        acc[fm][fn] = __builtin_amdgcn_wmma_f32_16x16x32_bf16(
            false, af[fm], false, bfv[fn], (short)0, acc[fm][fn], false, false);
  }
}

// ---------------------------------------------------------------------------
// GEMM1: u[b,i,c] = sum_{j<=i} M[i,j]*x1[b,j,c] + tri_b[i] + inputs[b,i,c]
// grid (4,4,64). Triangular: K-blocks above block-row diagonal are skipped.
// ---------------------------------------------------------------------------
__global__ __launch_bounds__(256) void gemm_tri_kernel(const __bf16* __restrict__ Mb,
                                                       const __bf16* __restrict__ x1t,
                                                       const float* __restrict__ x_in,
                                                       const float* __restrict__ tri_b,
                                                       float* __restrict__ u) {
  __shared__ __align__(128) __bf16 smem[NSTAGE * STAGE_ELEMS];
  const int b = blockIdx.z;
  const int m0 = blockIdx.y << 7;
  const int n0 = blockIdx.x << 7;
  const int nk = (m0 >> 5) + 4;  // only K-blocks touching/below the diagonal
  v8f acc[2][4];
#pragma unroll
  for (int i = 0; i < 2; ++i)
#pragma unroll
    for (int j = 0; j < 4; ++j) acc[i][j] = (v8f){0, 0, 0, 0, 0, 0, 0, 0};

  gemm_mainloop(Mb, x1t + ((size_t)b << 18), m0, n0, nk, smem, acc);

  const int tid = threadIdx.x, lane = tid & 31, wid = tid >> 5;
  const int wrow = wid & 3, wcol = wid >> 2, l16 = lane & 15, hi = lane >> 4;
#pragma unroll
  for (int fm = 0; fm < 2; ++fm)
#pragma unroll
    for (int fn = 0; fn < 4; ++fn) {
      const int col = n0 + (wcol << 6) + (fn << 4) + l16;
#pragma unroll
      for (int r = 0; r < 8; ++r) {
        const int row = m0 + (wrow << 5) + (fm << 4) + (hi << 3) + r;
        size_t idx = ((size_t)b << 18) + ((size_t)row << 9) + col;
        u[idx] = acc[fm][fn][r] + tri_b[row] + x_in[idx];
      }
    }
}

// ---------------------------------------------------------------------------
// GEMM2: H = gelu_exact(x2 @ d1_w^T + d1_b)  [32768x512] * [512x512]
// ---------------------------------------------------------------------------
__global__ __launch_bounds__(256) void gemm_mlp1_kernel(const __bf16* __restrict__ x2b,
                                                        const __bf16* __restrict__ W1b,
                                                        const float* __restrict__ b1,
                                                        __bf16* __restrict__ H) {
  __shared__ __align__(128) __bf16 smem[NSTAGE * STAGE_ELEMS];
  const int m0 = blockIdx.y << 7;
  const int n0 = blockIdx.x << 7;
  v8f acc[2][4];
#pragma unroll
  for (int i = 0; i < 2; ++i)
#pragma unroll
    for (int j = 0; j < 4; ++j) acc[i][j] = (v8f){0, 0, 0, 0, 0, 0, 0, 0};

  gemm_mainloop(x2b, W1b, m0, n0, TC / 32, smem, acc);

  const int tid = threadIdx.x, lane = tid & 31, wid = tid >> 5;
  const int wrow = wid & 3, wcol = wid >> 2, l16 = lane & 15, hi = lane >> 4;
#pragma unroll
  for (int fm = 0; fm < 2; ++fm)
#pragma unroll
    for (int fn = 0; fn < 4; ++fn) {
      const int col = n0 + (wcol << 6) + (fn << 4) + l16;
      const float bias = b1[col];
#pragma unroll
      for (int r = 0; r < 8; ++r) {
        const int row = m0 + (wrow << 5) + (fm << 4) + (hi << 3) + r;
        float v = acc[fm][fn][r] + bias;
        float g = 0.5f * v * (1.0f + erff(v * 0.70710678118654752f));
        H[((size_t)row << 9) + col] = (__bf16)g;
      }
    }
}

// ---------------------------------------------------------------------------
// GEMM3: out = x2 + H @ d2_w^T + d2_b   (out already holds x2 in f32)
// ---------------------------------------------------------------------------
__global__ __launch_bounds__(256) void gemm_mlp2_kernel(const __bf16* __restrict__ H,
                                                        const __bf16* __restrict__ W2b,
                                                        const float* __restrict__ b2,
                                                        float* __restrict__ out) {
  __shared__ __align__(128) __bf16 smem[NSTAGE * STAGE_ELEMS];
  const int m0 = blockIdx.y << 7;
  const int n0 = blockIdx.x << 7;
  v8f acc[2][4];
#pragma unroll
  for (int i = 0; i < 2; ++i)
#pragma unroll
    for (int j = 0; j < 4; ++j) acc[i][j] = (v8f){0, 0, 0, 0, 0, 0, 0, 0};

  gemm_mainloop(H, W2b, m0, n0, TC / 32, smem, acc);

  const int tid = threadIdx.x, lane = tid & 31, wid = tid >> 5;
  const int wrow = wid & 3, wcol = wid >> 2, l16 = lane & 15, hi = lane >> 4;
#pragma unroll
  for (int fm = 0; fm < 2; ++fm)
#pragma unroll
    for (int fn = 0; fn < 4; ++fn) {
      const int col = n0 + (wcol << 6) + (fn << 4) + l16;
      const float bias = b2[col];
#pragma unroll
      for (int r = 0; r < 8; ++r) {
        const int row = m0 + (wrow << 5) + (fm << 4) + (hi << 3) + r;
        size_t idx = ((size_t)row << 9) + col;
        out[idx] = out[idx] + acc[fm][fn][r] + bias;
      }
    }
}

// ---------------------------------------------------------------------------
extern "C" void kernel_launch(void* const* d_in, const int* in_sizes, int n_in,
                              void* d_out, int out_size, void* d_ws, size_t ws_size,
                              hipStream_t stream) {
  (void)in_sizes; (void)n_in; (void)out_size; (void)ws_size;

  const float* x_in = (const float*)d_in[0];
  const float* ln1w = (const float*)d_in[1];
  const float* ln1b = (const float*)d_in[2];
  const float* ln2w = (const float*)d_in[3];
  const float* ln2b = (const float*)d_in[4];
  const float* triM = (const float*)d_in[5];
  const float* trib = (const float*)d_in[6];
  const float* d1w  = (const float*)d_in[7];
  const float* d1b  = (const float*)d_in[8];
  const float* d2w  = (const float*)d_in[9];
  const float* d2b  = (const float*)d_in[10];
  float* out = (float*)d_out;

  // Workspace layout (~98 MB):
  char* w = (char*)d_ws;
  float* mu1   = (float*)w;
  float* rstd1 = mu1 + 64;
  float* mu2   = mu1 + 128;
  float* rstd2 = mu1 + 192;
  size_t off = 4096;
  __bf16* Mb  = (__bf16*)(w + off); off += (size_t)TT * TT * 2;   // 512 KB
  __bf16* W1b = (__bf16*)(w + off); off += (size_t)TC * TC * 2;   // 512 KB
  __bf16* W2b = (__bf16*)(w + off); off += (size_t)TC * TC * 2;   // 512 KB
  __bf16* x1t = (__bf16*)(w + off); off += NTOT * 2;              // 32 MB
  float*  u   = (float*)(w + off);  off += NTOT * 4;              // 64 MB
  __bf16* x2b = x1t;         // x1t dead after GEMM1
  __bf16* H   = (__bf16*)u;  // u dead after ln2_apply

  // 1. LN1 stats
  ln_stats_kernel<<<TB, 256, 0, stream>>>(x_in, mu1, rstd1);
  // 2. weight conversions
  tril_to_bf16_kernel<<<(TT * TT) / 256, 256, 0, stream>>>(triM, Mb);
  to_bf16_kernel<<<(TC * TC) / 256, 256, 0, stream>>>(d1w, W1b, TC * TC);
  to_bf16_kernel<<<(TC * TC) / 256, 256, 0, stream>>>(d2w, W2b, TC * TC);
  // 3. LN1 apply + transpose -> x1t (bf16, [b][c][t])
  ln1_apply_t_kernel<<<dim3(TC / 32, TT / 32, TB), dim3(32, 8), 0, stream>>>(
      x_in, ln1w, ln1b, mu1, rstd1, x1t);
  // 4. triangular time-mix + residual -> u (f32)
  gemm_tri_kernel<<<dim3(TC / 128, TT / 128, TB), 256, 0, stream>>>(Mb, x1t, x_in,
                                                                    trib, u);
  // 5. LN2 stats on u
  ln_stats_kernel<<<TB, 256, 0, stream>>>(u, mu2, rstd2);
  // 6. LN2 apply -> out (x2 f32) and x2b (bf16)
  ln2_apply_kernel<<<(unsigned)(NTOT / 256), 256, 0, stream>>>(u, ln2w, ln2b, mu2,
                                                               rstd2, out, x2b);
  // 7. MLP dense_1 + gelu -> H (bf16)
  gemm_mlp1_kernel<<<dim3(TC / 128, (TB * TT) / 128), 256, 0, stream>>>(x2b, W1b,
                                                                        d1b, H);
  // 8. MLP dense_2 + residual -> out
  gemm_mlp2_kernel<<<dim3(TC / 128, (TB * TT) / 128), 256, 0, stream>>>(H, W2b,
                                                                        d2b, out);
}
